// MatForward_4544075399860
// MI455X (gfx1250) — matrix operand order repaired and checked by reading, not compiled, not run on hardware
//
#include <hip/hip_runtime.h>

// ---- problem constants (from reference: B=8, T=2048, C=1024) ----
#define BB 8
#define TT 2048
#define CC 1024
#define MM (BB * TT)      // 16384 rows of the flattened [B*T, *] matrices
#define C2 (2 * CC)      // 2048

// ---- GEMM tiling ----
#define BM 128            // block tile M
#define BN 128            // block tile N
#define BK 64             // K-step staged in LDS
#define LDST 72           // LDS row stride in bf16 elems (64 + 8 pad = 144 B)

// ---- CDNA5 WMMA types ----
typedef __bf16 v16bf __attribute__((ext_vector_type(16)));
typedef float  v8f   __attribute__((ext_vector_type(8)));
typedef unsigned short us8  __attribute__((ext_vector_type(8)));
typedef unsigned short us16 __attribute__((ext_vector_type(16)));
typedef int v4i __attribute__((vector_size(4 * sizeof(int))));  // async-LDS builtin param

// ---- TDM descriptor vector types (6-arg builtin form) ----
typedef unsigned int tdm_g0_t __attribute__((vector_size(16)));  // uint32x4
typedef int          tdm_g1_t __attribute__((vector_size(32)));  // int32x8
typedef int          tdm_g2_t __attribute__((vector_size(16)));  // int32x4

// ---- gfx1250 feature probes ----
#if defined(__has_builtin)
#  if __has_builtin(__builtin_amdgcn_tensor_load_to_lds)
#    define HAVE_TDM 1
#  endif
#  if __has_builtin(__builtin_amdgcn_global_load_async_to_lds_b128)
#    define HAVE_ASYNC_LDS 1
#  endif
#endif
#ifndef HAVE_TDM
#  define HAVE_TDM 0
#endif
#ifndef HAVE_ASYNC_LDS
#  define HAVE_ASYNC_LDS 0
#endif

// ---------------------------------------------------------------------------
// TDM: one tensor_load_to_lds moves a whole 128x64 bf16 tile global->LDS,
// with hardware row padding into the LDST=72 layout. D# per cdna5_isa §8:
//  group0: count=1 | lds_addr | global_addr(57b) | type=2
//  group1: data_size=1(2B) | pad_enable, pad_interval=4 (32 DW = 128 B rows),
//          pad_amount=3 (4 DW = 16 B) | tensor_dim0=64 | tensor_dim1=128 |
//          tile_dim0=64 | tile_dim1=128 | tensor_dim0_stride=ldg
//  groups 2/3 (and trailing group): zero (2-D tile)
// ---------------------------------------------------------------------------
#if HAVE_TDM
__device__ __forceinline__ void tdm_load_tile(unsigned short* ldst,
                                              const unsigned short* gsrc,
                                              int ldg /*row stride, elems*/) {
  const unsigned long long ga = (unsigned long long)(const void*)gsrc;
  const unsigned int lo = (unsigned int)(unsigned long long)(void*)ldst;  // LDS byte addr

  tdm_g0_t g0;
  g0[0] = 1u;                                             // count=1 (valid, user)
  g0[1] = lo;                                             // lds_addr
  g0[2] = (unsigned int)(ga & 0xFFFFFFFFu);               // global_addr[31:0]
  g0[3] = (unsigned int)((ga >> 32) & 0x01FFFFFFu)        // global_addr[56:32]
        | (2u << 30);                                     // type=2 ("image")

  tdm_g1_t g1;
  g1[0] = (int)((1u << 16)      // data_size = 1 -> 2 bytes
              | (1u << 20)      // pad_enable
              | (4u << 22)      // pad_interval: 2^(4+1)=32 DWORDs (= 128 B row)
              | (3u << 25));    // pad_amount: 3+1 = 4 DWORDs (= 16 B)
  g1[1] = (int)(64u << 16);     // tensor_dim0[15:0]=64 (bits 63:48); abar=0
  g1[2] = (int)(128u << 16);    // tensor_dim0 hi=0; tensor_dim1[15:0]=128
  g1[3] = (int)(64u << 16);     // tensor_dim1 hi=0; tile_dim0=64
  g1[4] = (int)128u;            // tile_dim1=128; tile_dim2=0
  g1[5] = (int)ldg;             // tensor_dim0_stride[31:0] (elems)
  g1[6] = 0;                    // stride0 hi, tensor_dim1_stride lo
  g1[7] = 0;                    // tensor_dim1_stride hi
  tdm_g2_t gz4 = {0, 0, 0, 0};
  tdm_g1_t gz8 = {0, 0, 0, 0, 0, 0, 0, 0};
  __builtin_amdgcn_tensor_load_to_lds(g0, g1, gz4, gz4, gz8, 0);
}
#endif

// Copy 16 bytes global -> LDS: async DMA (ASYNCcnt) when available, else
// VGPR round-trip (global_load_b128 + ds_store_b128).
__device__ __forceinline__ void copy16(unsigned short* ldst,
                                       const unsigned short* gsrc) {
#if HAVE_ASYNC_LDS
  typedef __attribute__((address_space(1))) v4i gv4i;
  typedef __attribute__((address_space(3))) v4i lv4i;
  __builtin_amdgcn_global_load_async_to_lds_b128(
      (gv4i*)(unsigned long long)(const void*)gsrc,
      (lv4i*)(unsigned int)(unsigned long long)(void*)ldst,
      0, 0);
#else
  *(us8*)ldst = *(const us8*)gsrc;
#endif
}

// Per-lane staging fallback: 128 rows x 64 cols = 1024 x 16B chunks,
// 256 threads move 4 chunks each.
__device__ __forceinline__ void stage_tile_lanes(unsigned short* dst,
                                                 const unsigned short* src,
                                                 int row0, int k0, int ldg,
                                                 int tid) {
#pragma unroll
  for (int i = 0; i < 4; ++i) {
    const int id = tid + i * 256;
    const int r  = id >> 3;
    const int c  = (id & 7) * 8;
    copy16(dst + r * LDST + c, src + (size_t)(row0 + r) * ldg + (k0 + c));
  }
}

// Stage one tile (TDM: single descriptor issued by wave 0 only).
__device__ __forceinline__ void stage_tile(unsigned short* dst,
                                           const unsigned short* src,
                                           int row0, int k0, int ldg,
                                           int tid, int wave) {
#if HAVE_TDM
  if (wave == 0) tdm_load_tile(dst, src + (size_t)row0 * ldg + k0, ldg);
  (void)tid;
#else
  (void)wave;
  stage_tile_lanes(dst, src, row0, k0, ldg, tid);
#endif
}

// Wait for staged tiles to land in LDS, then workgroup barrier.
__device__ __forceinline__ void wait_stage(int wave) {
#if HAVE_TDM
  if (wave == 0) __builtin_amdgcn_s_wait_tensorcnt(0);
#elif HAVE_ASYNC_LDS
#  if __has_builtin(__builtin_amdgcn_s_wait_asynccnt)
  __builtin_amdgcn_s_wait_asynccnt(0);
#  else
  asm volatile("s_wait_asynccnt 0" ::: "memory");
#  endif
#endif
  (void)wave;
  __syncthreads();
}

__device__ __forceinline__ unsigned short f2bf(float f) {
  // round-to-nearest-even f32 -> bf16
  unsigned int u = __builtin_bit_cast(unsigned int, f);
  u += 0x7FFFu + ((u >> 16) & 1u);
  return (unsigned short)(u >> 16);
}

__device__ __forceinline__ v8f wmma_bf16(v16bf a, v16bf b, v8f c) {
  return __builtin_amdgcn_wmma_f32_16x16x32_bf16(false, a, false, b,
                                                 (short)0, c, false, false);
}

// ---------------------------------------------------------------------------
// LDS fragment loaders per CDNA5 ISA §7.12.2 (wave32):
// A (16x32 MxK): lane L -> row M=L%16; lanes 0-15 hold K 0..7 & 16..23,
//                lanes 16-31 hold K 8..15 & 24..31  (two 16B ds_load_b128).
// B (32x16 KxN): lane L -> col N=L%16; lanes 0-15 hold K 0..7 & 16..23,
//                lanes 16-31 hold K 8..15 & 24..31  (two 16B ds_load_b128).
// ---------------------------------------------------------------------------
__device__ __forceinline__ v16bf lds_frag_a(const unsigned short* sh,
                                            int m0, int ks, int lane) {
  const int m    = m0 + (lane & 15);
  const int koff = (lane < 16) ? 0 : 8;
  const unsigned short* p = sh + m * LDST + ks + koff;
  union { us8 h[2]; us16 v; } u;
  u.h[0] = *(const us8*)(p);
  u.h[1] = *(const us8*)(p + 16);
  return __builtin_bit_cast(v16bf, u.v);
}

__device__ __forceinline__ v16bf lds_frag_b(const unsigned short* sh,
                                            int n0, int ks, int lane) {
  const int n  = n0 + (lane & 15);
  const int kb = (lane < 16) ? 0 : 8;
  const unsigned short* p = sh + n * LDST + ks + kb;
  union { us8 h[2]; us16 v; } u;
  u.h[0] = *(const us8*)(p);
  u.h[1] = *(const us8*)(p + 16);
  return __builtin_bit_cast(v16bf, u.v);
}

// C/D 16x16 f32 tile: VGPR j -> row m0+j (lanes 0-15) / m0+8+j (lanes 16-31),
// col = n0 + lane%16. Optional fused elementwise multiply epilogue.
__device__ __forceinline__ void store_tile(float* __restrict__ C,
                                           const float* __restrict__ mul,
                                           int ldc, int m0, int n0,
                                           int lane, v8f acc) {
  const int col   = n0 + (lane & 15);
  const int rbase = m0 + ((lane >> 4) << 3);
#pragma unroll
  for (int j = 0; j < 8; ++j) {
    const size_t idx = (size_t)(rbase + j) * ldc + col;
    float v = acc[j];
    if (mul) v *= mul[idx];
    C[idx] = v;
  }
}

// ---------------------------------------------------------------------------
// LDS-staged, double-buffered bf16 WMMA GEMM:
//   C[M,N] (f32) = A[M,K] (bf16 row-major) @ W[N,K]^T (bf16 N-major/K-contig)
// Block = 256 threads = 8 waves; block tile 128x128, K-step 64.
// Waves arranged 4(M) x 2(N); wave tile 32x64 = 2x4 WMMA fragments.
// Tiles stream in via TDM (tensor_load_to_lds) or async-LDS DMA.
// ---------------------------------------------------------------------------
__global__ __launch_bounds__(256)
void gemm_bf16_wmma(const unsigned short* __restrict__ A,
                    const unsigned short* __restrict__ W,
                    float* __restrict__ C,
                    const float* __restrict__ mul,
                    int M, int N, int K) {
  __shared__ unsigned short shA[2][BM * LDST];
  __shared__ unsigned short shB[2][BN * LDST];

  const int tid  = threadIdx.x;
  const int lane = tid & 31;
  const int wave = tid >> 5;
  const int gm0  = blockIdx.y * BM;
  const int gn0  = blockIdx.x * BN;
  const int wm   = (wave & 3) * 32;   // wave M offset within block tile
  const int wn   = (wave >> 2) * 64;  // wave N offset within block tile

  v8f acc[2][4] = {};

  stage_tile(shA[0], A, gm0, 0, K, tid, wave);
  stage_tile(shB[0], W, gn0, 0, K, tid, wave);
  wait_stage(wave);

  int cur = 0;
  for (int k0 = 0; k0 < K; k0 += BK) {
    if (k0 + BK < K) {  // prefetch next K-tile into the other buffer
      stage_tile(shA[cur ^ 1], A, gm0, k0 + BK, K, tid, wave);
      stage_tile(shB[cur ^ 1], W, gn0, k0 + BK, K, tid, wave);
    }
#pragma unroll
    for (int ks = 0; ks < BK; ks += 32) {
      v16bf a0 = lds_frag_a(shA[cur], wm,      ks, lane);
      v16bf a1 = lds_frag_a(shA[cur], wm + 16, ks, lane);
#pragma unroll
      for (int j = 0; j < 4; ++j) {
        v16bf b = lds_frag_b(shB[cur], wn + j * 16, ks, lane);
        acc[0][j] = wmma_bf16(a0, b, acc[0][j]);
        acc[1][j] = wmma_bf16(a1, b, acc[1][j]);
      }
    }
    wait_stage(wave);
    cur ^= 1;
  }

#pragma unroll
  for (int j = 0; j < 4; ++j) {
    store_tile(C, mul, N, gm0 + wm,      gn0 + wn + j * 16, lane, acc[0][j]);
    store_tile(C, mul, N, gm0 + wm + 16, gn0 + wn + j * 16, lane, acc[1][j]);
  }
}

// ---------------------------------------------------------------------------
// f32 -> bf16 elementwise conversion (x, Wk, Wa, Wo are already [N,K]-major).
// ---------------------------------------------------------------------------
__global__ void cvt_bf16_kernel(const float* __restrict__ src,
                                unsigned short* __restrict__ dst, size_t n) {
  size_t i = (size_t)blockIdx.x * blockDim.x + threadIdx.x;
  if (i < n) dst[i] = f2bf(src[i]);
}

// Build combined conv weight Wfg[N=2C, K=2C] (N-major, K-contiguous):
//   K index = tap*C + cin (tap 0 = y[t-1], tap 1 = y[t])
__global__ void build_wfg_kernel(const float* __restrict__ Wf,
                                 const float* __restrict__ Wg,
                                 unsigned short* __restrict__ dst) {
  size_t i = (size_t)blockIdx.x * blockDim.x + threadIdx.x;
  const size_t total = (size_t)C2 * C2;
  if (i >= total) return;
  const int n   = (int)(i / C2);
  const int kk  = (int)(i % C2);
  const int tap = kk / CC;
  const int cin = kk % CC;
  const float* src = (n < CC) ? Wf : Wg;
  const int co = (n < CC) ? n : (n - CC);
  dst[i] = f2bf(src[(size_t)tap * CC * CC + (size_t)cin * CC + co]);
}

// ---------------------------------------------------------------------------
// Sequential scan over T per (b,c) channel:
//   scale_t = running max |cv_t|;  kv_t = prod_{i<=t} cv_i / scale_i
// Emits: re [M,2C] bf16 (view_as_real layout) and y2 [M,2C] bf16 =
// [ y_{t-1} | y_t ] shifted input for the fused K=2 causal convs.
// ---------------------------------------------------------------------------
__global__ void scan_kernel(const float* __restrict__ kmat,
                            const float* __restrict__ x,
                            unsigned short* __restrict__ re,
                            unsigned short* __restrict__ y2) {
  const int ch = blockIdx.x * blockDim.x + threadIdx.x;
  if (ch >= BB * CC) return;
  const int b = ch / CC;
  const int c = ch % CC;
  const size_t row0 = (size_t)b * TT;

  float m = 0.f, pr = 1.f, pi = 0.f;
  for (int t = 0; t < TT; ++t) {
    const size_t r = row0 + t;
    const float kr = kmat[r * C2 + 2 * c];
    const float ki = kmat[r * C2 + 2 * c + 1];
    const float a  = sqrtf(kr * kr + ki * ki);
    m = (t == 0) ? a : fmaxf(m, a);
    const float inv = 1.0f / m;
    const float cr = kr * inv, ci = ki * inv;
    const float nr = pr * cr - pi * ci;
    const float ni = pr * ci + pi * cr;
    pr = nr; pi = ni;
    re[r * C2 + 2 * c]     = f2bf(pr);
    re[r * C2 + 2 * c + 1] = f2bf(pi);
    const unsigned short yb = f2bf(x[r * CC + c] * m);
    y2[r * C2 + CC + c] = yb;                      // tap1: y_t for row t
    if (t + 1 < TT) y2[(r + 1) * C2 + c] = yb;     // tap0: y_t for row t+1
    if (t == 0)     y2[r * C2 + c] = 0;            // causal left-pad
  }
}

// h = tanh(f) * sigmoid(g), fg laid out [M, 2C] = [f | g]
__global__ void act_kernel(const float* __restrict__ fg,
                           unsigned short* __restrict__ h) {
  size_t i = (size_t)blockIdx.x * blockDim.x + threadIdx.x;
  if (i >= (size_t)MM * CC) return;
  const size_t m_ = i / CC;
  const int c = (int)(i % CC);
  const float f = fg[m_ * C2 + c];
  const float g = fg[m_ * C2 + CC + c];
  h[i] = f2bf(tanhf(f) * (1.f / (1.f + expf(-g))));
}

// ---------------------------------------------------------------------------
// Host-side orchestration
// ---------------------------------------------------------------------------
extern "C" void kernel_launch(void* const* d_in, const int* in_sizes, int n_in,
                              void* d_out, int out_size, void* d_ws, size_t ws_size,
                              hipStream_t stream) {
  (void)in_sizes; (void)n_in; (void)out_size; (void)ws_size;
  const float* x  = (const float*)d_in[0];  // [B,T,C]
  const float* Wk = (const float*)d_in[1];  // [2C,C]   (N-major, K-contig)
  const float* Wa = (const float*)d_in[2];  // [C,2C]   (N-major, K-contig)
  const float* Wf = (const float*)d_in[3];  // [2,C,C]
  const float* Wg = (const float*)d_in[4];  // [2,C,C]
  const float* Wo = (const float*)d_in[5];  // [C,C]    (N-major, K-contig)
  float* out = (float*)d_out;               // [B,T,C] f32

  unsigned char* ws = (unsigned char*)d_ws;
  // workspace layout (bytes); fg aliases k (dead after scan), h aliases re
  // (dead after gemm_a). Peak ~ 388 MB.
  const size_t o_xb   = 0;                                  // 32 MB bf16 x
  const size_t o_wk   = o_xb   + (size_t)MM * CC * 2;       //  4 MB
  const size_t o_wa   = o_wk   + (size_t)C2 * CC * 2;       //  4 MB
  const size_t o_wfg  = o_wa   + (size_t)CC * C2 * 2;       //  8 MB
  const size_t o_wo   = o_wfg  + (size_t)C2 * C2 * 2;       //  2 MB
  const size_t o_k    = o_wo   + (size_t)CC * CC * 2;       // 128 MB f32 k / fg
  const size_t o_re   = o_k    + (size_t)MM * C2 * 4;       //  64 MB bf16 re / h
  const size_t o_y2   = o_re   + (size_t)MM * C2 * 2;       //  64 MB bf16 y2
  const size_t o_out1 = o_y2   + (size_t)MM * C2 * 2;       //  64 MB f32 out1

  unsigned short* xb  = (unsigned short*)(ws + o_xb);
  unsigned short* wkb = (unsigned short*)(ws + o_wk);
  unsigned short* wab = (unsigned short*)(ws + o_wa);
  unsigned short* wfg = (unsigned short*)(ws + o_wfg);
  unsigned short* wob = (unsigned short*)(ws + o_wo);
  float*          kbf = (float*)(ws + o_k);
  unsigned short* reb = (unsigned short*)(ws + o_re);
  unsigned short* y2b = (unsigned short*)(ws + o_y2);
  float*          out1= (float*)(ws + o_out1);
  float*          fg  = kbf;                     // alias
  unsigned short* hb  = reb;                     // alias

  const int TPB = 256;
  auto blocks = [](size_t n) { return (unsigned)((n + 255) / 256); };

  // 1) convert activations + weights to bf16
  cvt_bf16_kernel<<<blocks((size_t)MM * CC), TPB, 0, stream>>>(x,  xb,  (size_t)MM * CC);
  cvt_bf16_kernel<<<blocks((size_t)C2 * CC), TPB, 0, stream>>>(Wk, wkb, (size_t)C2 * CC);
  cvt_bf16_kernel<<<blocks((size_t)CC * C2), TPB, 0, stream>>>(Wa, wab, (size_t)CC * C2);
  cvt_bf16_kernel<<<blocks((size_t)CC * CC), TPB, 0, stream>>>(Wo, wob, (size_t)CC * CC);
  build_wfg_kernel<<<blocks((size_t)C2 * C2), TPB, 0, stream>>>(Wf, Wg, wfg);

  // 2) k = x @ Wk.T          [16384,1024] x [1024,2048]
  gemm_bf16_wmma<<<dim3(C2 / BN, MM / BM), TPB, 0, stream>>>(
      xb, wkb, kbf, nullptr, MM, C2, CC);

  // 3) per-channel cummax / complex cumprod scan -> re, y2
  scan_kernel<<<(BB * CC) / TPB, TPB, 0, stream>>>(kbf, x, reb, y2b);

  // 4) out1 = re @ Wa.T      [16384,2048] x [2048,1024]
  gemm_bf16_wmma<<<dim3(CC / BN, MM / BM), TPB, 0, stream>>>(
      reb, wab, out1, nullptr, MM, CC, C2);

  // 5) fg = y2 @ Wfg.T       [16384,2048] x [2048,2048]  (fused f+g convs)
  gemm_bf16_wmma<<<dim3(C2 / BN, MM / BM), TPB, 0, stream>>>(
      y2b, wfg, fg, nullptr, MM, C2, C2);

  // 6) h = tanh(f) * sigmoid(g)
  act_kernel<<<blocks((size_t)MM * CC), TPB, 0, stream>>>(fg, hb);

  // 7) out = (h @ Wo.T) * out1   [16384,1024] x [1024,1024], fused epilogue
  gemm_bf16_wmma<<<dim3(CC / BN, MM / BM), TPB, 0, stream>>>(
      hb, wob, out, out1, MM, CC, CC);
}
